// Faster_R_CNN_30219389895050
// MI455X (gfx1250) — matrix operand order repaired and checked by reading, not compiled, not run on hardware
//
#include <hip/hip_runtime.h>

typedef __attribute__((ext_vector_type(16))) __bf16 v16bf;
typedef __attribute__((ext_vector_type(8)))  float  v8f;

#define NEGF (-__builtin_inff())

__device__ __forceinline__ unsigned short f2bf(float f) {
  unsigned int u = __float_as_uint(f);
  unsigned int r = (u + 0x7FFFu + ((u >> 16) & 1u)) >> 16;  // RNE
  return (unsigned short)r;
}

// Fragment build + 4 WMMAs for one K-step of a wave's 16x64 strip.
// A 16x32: lane&15 = M row, half h = lane>>4; elems 0-7 = K 8h..8h+7,
//          elems 8-15 = K 16+8h..16+8h+7 (contiguous -> ds_load_b128 x2).
// B 32x16: lane&15 = N col, half h; elems 0-7 = K 8h..8h+7, elems 8-15 = K 16+8h..16+8h+7.
// C/D:     n = lane&15, VGPR r -> m = r + 8h.
__device__ __forceinline__ void frag_mma4(
    const unsigned short (*As)[40], const unsigned short (*Bs)[40],
    int arow, int h, int l15, v8f acc[4]) {
  union { v16bf v; uint4 q[2]; } af;
  af.q[0] = *(const uint4*)&As[arow][8 * h];
  af.q[1] = *(const uint4*)&As[arow][16 + 8 * h];
  #pragma unroll
  for (int t = 0; t < 4; ++t) {
    union { v16bf v; uint4 q[2]; } bf;
    const int brow = t * 16 + l15;
    bf.q[0] = *(const uint4*)&Bs[brow][8 * h];
    bf.q[1] = *(const uint4*)&Bs[brow][16 + 8 * h];
    acc[t] = __builtin_amdgcn_wmma_f32_16x16x32_bf16(
        false, af.v, false, bf.v, (short)0, acc[t], false, false);
  }
}

// ---------------------------------------------------------------------------
// C[M,N] = A[M,K] * W[N,K]^T + bias[N]  (optional ReLU).  Requires K % 32 == 0.
// 256 threads = 8 waves. Block tile: 128(M) x 64(N), BK=32.
// Each wave owns a 16x64 strip: one A fragment reused against 4 B fragments
// -> 4 x v_wmma_f32_16x16x32_bf16 per wave per K-step.
// LDS rows padded to 40 ushorts (80B, a 16B multiple) so fragments are
// aligned ds_load_b128's. The K-loop is manually versioned on a wave-uniform
// "interior" predicate so interior blocks get a pure
// global_load_b128 -> RNE-cvt -> ds_store_b64 fill with no exec masking.
// ---------------------------------------------------------------------------
__global__ __launch_bounds__(256) void gemm_wmma_bf16(
    const float* __restrict__ A, const float* __restrict__ W,
    const float* __restrict__ bias, float* __restrict__ C,
    int M, int N, int K, int relu)
{
  __shared__ unsigned short As[128][40];
  __shared__ unsigned short Bs[64][40];

  const int tid  = threadIdx.x;
  const int lane = tid & 31;
  const int wv   = tid >> 5;      // wave 0..7 -> 16-row strip
  const int h    = lane >> 4;     // lane half
  const int l15  = lane & 15;
  const long m0  = (long)blockIdx.y * 128;
  const long n0  = (long)blockIdx.x * 64;

  v8f acc[4] = {{}, {}, {}, {}};

  const int arow = wv * 16 + l15;
  const bool interior = (m0 + 128 <= M) && (n0 + 64 <= N);

  if (interior) {
    // -------- fast path: no guards anywhere in the fill --------
    for (int k0 = 0; k0 < K; k0 += 32) {
      #pragma unroll
      for (int i = 0; i < 4; ++i) {           // A: 128x32, 4 float4 / thread
        int e = i * 256 + tid;
        int r = e >> 3, c = (e & 7) * 4;
        float4 q = *(const float4*)(A + (m0 + r) * K + k0 + c);
        ushort4 hq;
        hq.x = f2bf(q.x); hq.y = f2bf(q.y); hq.z = f2bf(q.z); hq.w = f2bf(q.w);
        *(ushort4*)&As[r][c] = hq;
      }
      #pragma unroll
      for (int i = 0; i < 2; ++i) {           // B: 64x32, 2 float4 / thread
        int e = i * 256 + tid;
        int r = e >> 3, c = (e & 7) * 4;
        float4 q = *(const float4*)(W + (n0 + r) * K + k0 + c);
        ushort4 hq;
        hq.x = f2bf(q.x); hq.y = f2bf(q.y); hq.z = f2bf(q.z); hq.w = f2bf(q.w);
        *(ushort4*)&Bs[r][c] = hq;
      }
      __syncthreads();
      frag_mma4(As, Bs, arow, h, l15, acc);
      __syncthreads();
    }
  } else {
    // -------- edge path: clamped row + value select --------
    for (int k0 = 0; k0 < K; k0 += 32) {
      #pragma unroll
      for (int i = 0; i < 4; ++i) {
        int e = i * 256 + tid;
        int r = e >> 3, c = (e & 7) * 4;
        long m  = m0 + r;
        long mc = (m < (long)M - 1) ? m : (long)M - 1;
        float4 q = *(const float4*)(A + mc * K + k0 + c);
        bool ok = (m < M);
        ushort4 hq;
        hq.x = ok ? f2bf(q.x) : (unsigned short)0;
        hq.y = ok ? f2bf(q.y) : (unsigned short)0;
        hq.z = ok ? f2bf(q.z) : (unsigned short)0;
        hq.w = ok ? f2bf(q.w) : (unsigned short)0;
        *(ushort4*)&As[r][c] = hq;
      }
      #pragma unroll
      for (int i = 0; i < 2; ++i) {
        int e = i * 256 + tid;
        int r = e >> 3, c = (e & 7) * 4;
        long n  = n0 + r;
        long nc = (n < (long)N - 1) ? n : (long)N - 1;
        float4 q = *(const float4*)(W + nc * K + k0 + c);
        bool ok = (n < N);
        ushort4 hq;
        hq.x = ok ? f2bf(q.x) : (unsigned short)0;
        hq.y = ok ? f2bf(q.y) : (unsigned short)0;
        hq.z = ok ? f2bf(q.z) : (unsigned short)0;
        hq.w = ok ? f2bf(q.w) : (unsigned short)0;
        *(ushort4*)&Bs[r][c] = hq;
      }
      __syncthreads();
      frag_mma4(As, Bs, arow, h, l15, acc);
      __syncthreads();
    }
  }

  // ---- store ----
  const int mloc = wv * 16 + 8 * h;
  #pragma unroll
  for (int t = 0; t < 4; ++t) {
    long n = n0 + t * 16 + l15;
    if (n < N) {
      float bv = bias[n];
      #pragma unroll
      for (int r = 0; r < 8; ++r) {
        long m = m0 + mloc + r;
        if (m < M) {
          float v = acc[t][r] + bv;
          C[m * N + n] = relu ? fmaxf(v, 0.f) : v;
        }
      }
    }
  }
}

// ---------------------------------------------------------------------------
// im2col for the 3x3 SAME conv: features (512,25,25) -> col (625, 4608)
// k = ic*9 + ky*3 + kx  (matches OIHW weight flattening)
// ---------------------------------------------------------------------------
__global__ void im2col_k(const float* __restrict__ F, float* __restrict__ col) {
  int idx = blockIdx.x * 256 + threadIdx.x;
  if (idx >= 625 * 4608) return;
  int p = idx / 4608, k = idx - p * 4608;
  int ic = k / 9, r = k - ic * 9;
  int ky = r / 3, kx = r - ky * 3;
  int y = p / 25, x = p - y * 25;
  int iy = y + ky - 1, ix = x + kx - 1;
  float v = 0.f;
  if (iy >= 0 && iy < 25 && ix >= 0 && ix < 25) v = F[ic * 625 + iy * 25 + ix];
  col[idx] = v;
}

// ---------------------------------------------------------------------------
// Per-anchor softmax prob, anchor gen, box decode, clip, min-size filter.
// Anchor i = p*9 + a; a = ratio*3 + size; stride 32, H=W=800.
// ---------------------------------------------------------------------------
__global__ void decode_k(const float* __restrict__ clspc, const float* __restrict__ regpc,
                         float* __restrict__ boxes, float* __restrict__ score) {
  int i = blockIdx.x * 256 + threadIdx.x;
  if (i >= 5625) return;
  int p = i / 9, a = i - p * 9;
  int ri = a / 3, si = a - ri * 3;
  const float sizes[3]  = {128.f, 256.f, 512.f};
  const float ratios[3] = {0.5f, 1.f, 2.f};
  float hr = sqrtf(ratios[ri]);
  float wr = 1.f / hr;
  float wsz = wr * sizes[si], hsz = hr * sizes[si];
  float sx = (float)(p % 25) * 32.f;
  float sy = (float)(p / 25) * 32.f;
  float ax1 = sx + roundf(-wsz * 0.5f), ay1 = sy + roundf(-hsz * 0.5f);
  float ax2 = sx + roundf( wsz * 0.5f), ay2 = sy + roundf( hsz * 0.5f);
  float axc = (ax1 + ax2) * 0.5f, ayc = (ay1 + ay2) * 0.5f;
  float aw = fmaxf(ax2 - ax1, 1e-6f), ah = fmaxf(ay2 - ay1, 1e-6f);
  float tx = regpc[p * 36 + a * 4 + 0];
  float ty = regpc[p * 36 + a * 4 + 1];
  float tw = regpc[p * 36 + a * 4 + 2];
  float th = regpc[p * 36 + a * 4 + 3];
  float px = tx * aw + axc, py = ty * ah + ayc;
  float pw = aw * fminf(expf(tw), 1e6f);
  float ph = ah * fminf(expf(th), 1e6f);
  float x1 = fminf(fmaxf(px - 0.5f * pw, 0.f), 799.f);
  float y1 = fminf(fmaxf(py - 0.5f * ph, 0.f), 799.f);
  float x2 = fminf(fmaxf(px + 0.5f * pw, 0.f), 799.f);
  float y2 = fminf(fmaxf(py + 0.5f * ph, 0.f), 799.f);
  boxes[i * 4 + 0] = x1; boxes[i * 4 + 1] = y1;
  boxes[i * 4 + 2] = x2; boxes[i * 4 + 3] = y2;
  float l0 = clspc[p * 18 + 2 * a], l1 = clspc[p * 18 + 2 * a + 1];
  float prob = 1.f / (1.f + expf(l0 - l1));
  float wsd = fmaxf(x2 - x1, 0.f), hsd = fmaxf(y2 - y1, 0.f);
  score[i] = (wsd >= 1.f && hsd >= 1.f) ? prob : NEGF;
}

// ---------------------------------------------------------------------------
// Greedy NMS: 1000 sequential rounds of (parallel argmax over 5625) + IoU
// suppress. Single workgroup; argmax tie-breaks to lowest index (jnp.argmax).
// Writes proposals (boxes[keep]*kv) and valid flags straight into d_out.
// ---------------------------------------------------------------------------
__global__ __launch_bounds__(256) void nms_k(
    const float* __restrict__ boxes, float* __restrict__ score,
    float* __restrict__ areas, float* __restrict__ prop,
    float* __restrict__ validf) {
  __shared__ float sv[256];
  __shared__ int   si[256];
  __shared__ float jb[4];
  __shared__ float jarea;
  __shared__ int   jvalid;
  int tid = threadIdx.x;

  for (int i = tid; i < 5625; i += 256) {
    float x1 = boxes[i*4], y1 = boxes[i*4+1], x2 = boxes[i*4+2], y2 = boxes[i*4+3];
    areas[i] = (x2 - x1) * (y2 - y1);
  }
  __syncthreads();

  for (int it = 0; it < 1000; ++it) {
    float bv = NEGF; int bi = 1 << 30;
    for (int i = tid; i < 5625; i += 256) {
      float v = score[i];
      if (v > bv || (v == bv && i < bi)) { bv = v; bi = i; }
    }
    sv[tid] = bv; si[tid] = bi;
    __syncthreads();
    for (int s = 128; s > 0; s >>= 1) {
      if (tid < s) {
        if (sv[tid + s] > sv[tid] ||
            (sv[tid + s] == sv[tid] && si[tid + s] < si[tid])) {
          sv[tid] = sv[tid + s]; si[tid] = si[tid + s];
        }
      }
      __syncthreads();
    }
    if (tid == 0) {
      int j = si[0];
      int valid = (sv[0] > NEGF) ? 1 : 0;
      jvalid = valid;
      jb[0] = boxes[j*4];   jb[1] = boxes[j*4+1];
      jb[2] = boxes[j*4+2]; jb[3] = boxes[j*4+3];
      jarea = areas[j];
      int kj = valid ? j : 0;
      float kvf = valid ? 1.f : 0.f;
      prop[it*4+0] = boxes[kj*4+0] * kvf;
      prop[it*4+1] = boxes[kj*4+1] * kvf;
      prop[it*4+2] = boxes[kj*4+2] * kvf;
      prop[it*4+3] = boxes[kj*4+3] * kvf;
      validf[it] = kvf;
      score[j] = NEGF;
    }
    __syncthreads();
    if (jvalid) {
      float jx1 = jb[0], jy1 = jb[1], jx2 = jb[2], jy2 = jb[3], ja = jarea;
      for (int i = tid; i < 5625; i += 256) {
        float iw = fmaxf(fminf(boxes[i*4+2], jx2) - fmaxf(boxes[i*4+0], jx1), 0.f);
        float ih = fmaxf(fminf(boxes[i*4+3], jy2) - fmaxf(boxes[i*4+1], jy1), 0.f);
        float inter = iw * ih;
        float iou = inter / fmaxf(areas[i] + ja - inter, 1e-12f);
        if (iou > 0.7f) score[i] = NEGF;
      }
    }
    __syncthreads();
  }
}

// ---------------------------------------------------------------------------
// ROI max-pool (7x7, quantized, capped at 6x6 samples per bin like the
// reference's ROI_PAD loop). One thread per output element; the 1.28 MB
// feature map lives in L2. Output layout (R, C, 7, 7) -> (R, 25088).
// ---------------------------------------------------------------------------
__global__ void roipool_k(const float* __restrict__ feat,
                          const float* __restrict__ prop,
                          const float* __restrict__ validf,
                          float* __restrict__ out) {
  long idx = (long)blockIdx.x * 256 + threadIdx.x;
  const long total = 1000L * 512 * 49;
  if (idx >= total) return;
  int r   = (int)(idx / (512 * 49));
  int rem = (int)(idx - (long)r * 512 * 49);
  int c   = rem / 49;
  int bin = rem - c * 49;
  int ph = bin / 7, pw = bin - ph * 7;

  const float scale = 1.f / 32.f;
  int x1i = (int)roundf(prop[r*4+0] * scale);
  int y1i = (int)roundf(prop[r*4+1] * scale);
  int x2i = (int)roundf(prop[r*4+2] * scale);
  int y2i = (int)roundf(prop[r*4+3] * scale);
  float rw = (float)max(x2i - x1i + 1, 1);
  float rh = (float)max(y2i - y1i + 1, 1);
  float bw = rw / 7.f, bh = rh / 7.f;
  int wstart = min(max((int)floorf((float)pw * bw) + x1i, 0), 25);
  int wend   = min(max((int)ceilf((float)(pw + 1) * bw) + x1i, 0), 25);
  int hstart = min(max((int)floorf((float)ph * bh) + y1i, 0), 25);
  int hend   = min(max((int)ceilf((float)(ph + 1) * bh) + y1i, 0), 25);

  float m = NEGF;
  for (int dh = 0; dh < 6; ++dh) {
    int hh = hstart + dh;
    if (hh >= hend) continue;
    int hc = min(hh, 24);
    for (int dw = 0; dw < 6; ++dw) {
      int w = wstart + dw;
      if (w >= wend) continue;
      int wc = min(w, 24);
      m = fmaxf(m, feat[c * 625 + hc * 25 + wc]);
    }
  }
  if (m == NEGF) m = 0.f;
  out[(long)r * 25088 + c * 49 + bin] = m * validf[r];
}

// ---------------------------------------------------------------------------
extern "C" void kernel_launch(void* const* d_in, const int* in_sizes, int n_in,
                              void* d_out, int out_size, void* d_ws, size_t ws_size,
                              hipStream_t stream) {
  (void)in_sizes; (void)n_in; (void)out_size; (void)ws_size;

  const float* features = (const float*)d_in[0];   // (1,512,25,25)
  const float* conv_w   = (const float*)d_in[1];   // (512,512,3,3)
  const float* conv_b   = (const float*)d_in[2];
  const float* cls_w    = (const float*)d_in[3];   // (18,512,1,1)
  const float* cls_b    = (const float*)d_in[4];
  const float* reg_w    = (const float*)d_in[5];   // (36,512,1,1)
  const float* reg_b    = (const float*)d_in[6];
  const float* fc1_w    = (const float*)d_in[7];   // (4096,25088)
  const float* fc1_b    = (const float*)d_in[8];
  const float* fc2_w    = (const float*)d_in[9];   // (4096,4096)
  const float* fc2_b    = (const float*)d_in[10];
  const float* csw      = (const float*)d_in[11];  // (21,4096)
  const float* csb      = (const float*)d_in[12];
  const float* bbw      = (const float*)d_in[13];  // (4,4096)
  const float* bbb      = (const float*)d_in[14];
  // d_in[15]=H(800), d_in[16]=W(800): fixed by the reference setup.

  float* ws    = (float*)d_ws;
  float* col   = ws;                 // 625*4608 = 2,880,000
  float* xpc   = col + 2880000;      // 625*512  =   320,000
  float* clspc = xpc + 320000;       // 625*18 (pad 11264)
  float* regpc = clspc + 11264;      // 625*36 (pad 22528)
  float* boxes = regpc + 22528;      // 5625*4 (pad 22528)
  float* score = boxes + 22528;      // 5625 (pad 5632)
  float* areas = score + 5632;       // 5625 (pad 5632)
  float* feats = areas + 5632;       // 1000*25088 = 25,088,000
  float* h1    = feats + 25088000;   // 1000*4096
  float* h2    = h1 + 4096000;       // 1000*4096

  float* outp   = (float*)d_out;
  float* prop   = outp;          // proposals: 1000*4
  float* valf   = outp + 4000;   // valid:     1000
  float* logits = outp + 5000;   // logits:    1000*21
  float* deltas = outp + 26000;  // deltas:    1000*4

  dim3 blk(256);

  // RPN trunk: conv3x3(+ReLU) and 1x1 heads as WMMA GEMMs over im2col / pixel-major acts.
  im2col_k<<<(625 * 4608 + 255) / 256, blk, 0, stream>>>(features, col);
  gemm_wmma_bf16<<<dim3(512 / 64, (625 + 127) / 128), blk, 0, stream>>>(
      col, conv_w, conv_b, xpc, 625, 512, 4608, 1);
  gemm_wmma_bf16<<<dim3(1, (625 + 127) / 128), blk, 0, stream>>>(
      xpc, cls_w, cls_b, clspc, 625, 18, 512, 0);
  gemm_wmma_bf16<<<dim3(1, (625 + 127) / 128), blk, 0, stream>>>(
      xpc, reg_w, reg_b, regpc, 625, 36, 512, 0);

  // Proposal generation + NMS (+ proposals/valid to d_out) + ROI pooling.
  decode_k<<<(5625 + 255) / 256, blk, 0, stream>>>(clspc, regpc, boxes, score);
  nms_k<<<1, blk, 0, stream>>>(boxes, score, areas, prop, valf);
  roipool_k<<<(1000 * 512 * 49 + 255) / 256, blk, 0, stream>>>(
      features, prop, valf, feats);

  // Detection head: the dominant GEMMs (fc1 streams 411 MB of weights).
  gemm_wmma_bf16<<<dim3(4096 / 64, (1000 + 127) / 128), blk, 0, stream>>>(
      feats, fc1_w, fc1_b, h1, 1000, 4096, 25088, 0);
  gemm_wmma_bf16<<<dim3(4096 / 64, (1000 + 127) / 128), blk, 0, stream>>>(
      h1, fc2_w, fc2_b, h2, 1000, 4096, 4096, 0);
  gemm_wmma_bf16<<<dim3(1, (1000 + 127) / 128), blk, 0, stream>>>(
      h2, csw, csb, logits, 1000, 21, 4096, 0);
  gemm_wmma_bf16<<<dim3(1, (1000 + 127) / 128), blk, 0, stream>>>(
      h2, bbw, bbb, deltas, 1000, 4, 4096, 0);
}
